// HGPMamba_51496657879072
// MI455X (gfx1250) — matrix operand order repaired and checked by reading, not compiled, not run on hardware
//
#include <hip/hip_runtime.h>

typedef __attribute__((ext_vector_type(16))) __bf16 v16bf;
typedef __attribute__((ext_vector_type(8)))  __bf16 v8bf;
typedef __attribute__((ext_vector_type(8)))  float  v8f;

#define DMODEL 256
#define DINNER 512
#define CHUNK  64     // timesteps per scan chunk (L=4096 -> 64 chunks, L=8192 -> 128)

// ---------------------------------------------------------------------------
// FAST WMMA bf16 GEMM: requires K%32==0, M%64==0, N%64==0.
// C[M,N] = act(A[M,K] * Bt[N,K]^T + bias) (+ res). One wave: 16(M) x 64(N).
// ---------------------------------------------------------------------------
template<int ACT, int BIAS, int RES>
__global__ __launch_bounds__(128)
void gemm_bf16_fast(const __bf16* __restrict__ A, const __bf16* __restrict__ Bt,
                    float* __restrict__ C, const float* __restrict__ bias,
                    const float* __restrict__ res, int M, int N, int K)
{
    const int lane  = threadIdx.x & 31;
    const int wave  = threadIdx.x >> 5;
    const int tileM = blockIdx.x * 64 + wave * 16;
    const int tileN = blockIdx.y * 64;
    const int hl    = lane >> 4;
    const int l15   = lane & 15;

    const __bf16* pa = A + (size_t)(tileM + l15) * K + hl * 8;
    const __bf16* pb[4];
#pragma unroll
    for (int nt = 0; nt < 4; ++nt)
        pb[nt] = Bt + (size_t)(tileN + nt * 16 + l15) * K + hl * 8;

    v8f acc[4] = {};

    for (int k0 = 0; k0 < K; k0 += 32) {
        v8bf alo = *(const v8bf*)(pa);
        v8bf ahi = *(const v8bf*)(pa + 16);
        v16bf a;
#pragma unroll
        for (int e = 0; e < 8; ++e) { a[e] = alo[e]; a[e + 8] = ahi[e]; }
        __builtin_prefetch(pa + 32, 0, 1);
        pa += 32;
#pragma unroll
        for (int nt = 0; nt < 4; ++nt) {
            v8bf blo = *(const v8bf*)(pb[nt]);
            v8bf bhi = *(const v8bf*)(pb[nt] + 16);
            pb[nt] += 32;
            v16bf b;
#pragma unroll
            for (int e = 0; e < 8; ++e) { b[e] = blo[e]; b[e + 8] = bhi[e]; }
            acc[nt] = __builtin_amdgcn_wmma_f32_16x16x32_bf16(
                false, a, false, b, (short)0, acc[nt], false, false);
        }
    }

#pragma unroll
    for (int nt = 0; nt < 4; ++nt) {
        const int col = tileN + nt * 16 + l15;
        const float bv = BIAS ? bias[col] : 0.0f;
#pragma unroll
        for (int v = 0; v < 8; ++v) {
            const int m = tileM + v + hl * 8;
            float val = acc[nt][v] + bv;
            if (ACT == 1) val = fmaxf(val, 0.0f);
            if (RES) val += res[(size_t)m * N + col];
            C[(size_t)m * N + col] = val;
        }
    }
}

// ---------------------------------------------------------------------------
// EDGE WMMA GEMM: any M/N/K (only used for K=50, N=48, K=16 small GEMMs)
// ---------------------------------------------------------------------------
__global__ __launch_bounds__(128)
void gemm_bf16_edge(const __bf16* __restrict__ A, const __bf16* __restrict__ Bt,
                    float* __restrict__ C, const float* __restrict__ bias,
                    const float* __restrict__ res, int M, int N, int K, int act)
{
    const int lane  = threadIdx.x & 31;
    const int wave  = threadIdx.x >> 5;
    const int tileM = blockIdx.x * 64 + wave * 16;
    const int tileN = blockIdx.y * 64;
    const int hl    = lane >> 4;
    const int l15   = lane & 15;
    const int rowA  = tileM + l15;

    v8f acc[4] = {};

    for (int k0 = 0; k0 < K; k0 += 32) {
        v16bf a;
        const int ka = k0 + hl * 8;
#pragma unroll
        for (int e = 0; e < 8; ++e) {
            int k1 = ka + e, k2 = ka + 16 + e;
            a[e]     = (rowA < M && k1 < K) ? A[(size_t)rowA * K + k1] : (__bf16)0.0f;
            a[e + 8] = (rowA < M && k2 < K) ? A[(size_t)rowA * K + k2] : (__bf16)0.0f;
        }
        const int kb = k0 + hl * 8;
#pragma unroll
        for (int nt = 0; nt < 4; ++nt) {
            const int col = tileN + nt * 16 + l15;
            v16bf b;
#pragma unroll
            for (int e = 0; e < 16; ++e) {
                int k = kb + e + (e >= 8 ? 8 : 0);
                b[e] = (col < N && k < K) ? Bt[(size_t)col * K + k] : (__bf16)0.0f;
            }
            acc[nt] = __builtin_amdgcn_wmma_f32_16x16x32_bf16(
                false, a, false, b, (short)0, acc[nt], false, false);
        }
    }

#pragma unroll
    for (int nt = 0; nt < 4; ++nt) {
        const int col = tileN + nt * 16 + l15;
        if (col >= N) continue;
        float bv = bias ? bias[col] : 0.0f;
#pragma unroll
        for (int v = 0; v < 8; ++v) {
            int m = tileM + v + hl * 8;
            if (m >= M) continue;
            float val = acc[nt][v] + bv;
            if (act == 1) val = fmaxf(val, 0.0f);
            if (res) val += res[(size_t)m * N + col];
            C[(size_t)m * N + col] = val;
        }
    }
}

// ---------------------------------------------------------------------------
// fp32 -> bf16 convert with leading-dim / column-slice support
// ---------------------------------------------------------------------------
__global__ void f2bf_slice(const float* __restrict__ src, __bf16* __restrict__ dst,
                           int rows, int cols, int ld, int coff)
{
    int idx = blockIdx.x * 256 + threadIdx.x;
    if (idx >= rows * cols) return;
    int c = idx % cols, r = idx / cols;
    dst[idx] = (__bf16)src[(size_t)r * ld + coff + c];
}

// fp32 (K x N) -> bf16 (N x K) transpose (weights)
__global__ void transpose_f2bf(const float* __restrict__ src, __bf16* __restrict__ dst,
                               int K, int N)
{
    __shared__ float tile[32][33];
    int k0 = blockIdx.x * 32, n0 = blockIdx.y * 32;
    int tx = threadIdx.x, ty = threadIdx.y;
    int k = k0 + ty, n = n0 + tx;
    tile[ty][tx] = (k < K && n < N) ? src[(size_t)k * N + n] : 0.0f;
    __syncthreads();
    int n2 = n0 + ty, k2 = k0 + tx;
    if (n2 < N && k2 < K) dst[(size_t)n2 * K + k2] = (__bf16)tile[tx][ty];
}

// ---------------------------------------------------------------------------
// LayerNorm over 256 features, one row per block
// ---------------------------------------------------------------------------
__global__ __launch_bounds__(256)
void layernorm256(const float* __restrict__ x, const float* __restrict__ g,
                  const float* __restrict__ b, float* __restrict__ out)
{
    __shared__ float red[256];
    int r = blockIdx.x, c = threadIdx.x;
    float v = x[(size_t)r * 256 + c];
    red[c] = v; __syncthreads();
    for (int s = 128; s > 0; s >>= 1) { if (c < s) red[c] += red[c + s]; __syncthreads(); }
    float mean = red[0] * (1.0f / 256.0f); __syncthreads();
    float dv = v - mean;
    red[c] = dv * dv; __syncthreads();
    for (int s = 128; s > 0; s >>= 1) { if (c < s) red[c] += red[c + s]; __syncthreads(); }
    float var = red[0] * (1.0f / 256.0f);
    out[(size_t)r * 256 + c] = dv * rsqrtf(var + 1e-5f) * g[c] + b[c];
}

// ---------------------------------------------------------------------------
// Causal depthwise conv1d (k=4) + SiLU; out: fp32 + bf16 (L,512)
// ---------------------------------------------------------------------------
__global__ void conv1d_silu(const float* __restrict__ xin, int ld, int coff,
                            const float* __restrict__ w, const float* __restrict__ b,
                            float* __restrict__ u, __bf16* __restrict__ ubf, int Lq)
{
    int idx = blockIdx.x * 256 + threadIdx.x;
    if (idx >= Lq * DINNER) return;
    int d = idx & (DINNER - 1);
    int t = idx >> 9;
    float acc = b[d];
#pragma unroll
    for (int j = 0; j < 4; ++j) {
        int tt = t - 3 + j;
        if (tt >= 0) acc += w[d * 4 + j] * xin[(size_t)tt * ld + coff + d];
    }
    float s = acc / (1.0f + __expf(-acc));
    u[idx] = s;
    ubf[idx] = (__bf16)s;
}

// ---------------------------------------------------------------------------
// Chunked parallel selective scan. Recurrence h_t = a_t h_{t-1} + b_t with
// a_t = exp(softplus(delta+bias)*A), b_t = dt*u*B_t.
// Phase 1: per-chunk local (P = prod a, S = local final h), 1M-thread parallel.
// Phase 2: sequential carry composition over <=128 chunks (8192 threads).
// Phase 3: re-run each chunk from its correct incoming state, emit y.
// ---------------------------------------------------------------------------
__global__ __launch_bounds__(256)
void sscan_phase1(const float* __restrict__ delta, const float* __restrict__ dt_bias,
                  const float* __restrict__ u, const float* __restrict__ xdbl,
                  const float* __restrict__ A_log,
                  float* __restrict__ Pc, float* __restrict__ Sc)
{
    const int n = threadIdx.x & 15;
    const int d = blockIdx.x * 16 + (threadIdx.x >> 4);
    const int c = blockIdx.y;
    const int t0 = c * CHUNK;
    const float A    = -__expf(A_log[d * 16 + n]);
    const float bias = dt_bias[d];
    float P = 1.0f, h = 0.0f;
    for (int t = t0; t < t0 + CHUNK; ++t) {
        float draw = delta[(size_t)t * DINNER + d] + bias;
        float dt = (draw > 20.0f) ? draw : log1pf(__expf(draw));
        float a = __expf(dt * A);
        float b = dt * u[(size_t)t * DINNER + d] * xdbl[(size_t)t * 48 + 16 + n];
        P *= a;
        h = a * h + b;
    }
    const size_t i = (size_t)c * 8192 + d * 16 + n;
    Pc[i] = P;
    Sc[i] = h;
}

__global__ __launch_bounds__(256)
void sscan_carry(const float* __restrict__ Pc, const float* __restrict__ Sc,
                 float* __restrict__ Hin, int C)
{
    const int i = blockIdx.x * 256 + threadIdx.x;   // 0..8191
    float carry = 0.0f;
    for (int c = 0; c < C; ++c) {
        Hin[(size_t)c * 8192 + i] = carry;
        carry = Pc[(size_t)c * 8192 + i] * carry + Sc[(size_t)c * 8192 + i];
    }
}

__global__ __launch_bounds__(256)
void sscan_phase3(const float* __restrict__ delta, const float* __restrict__ dt_bias,
                  const float* __restrict__ u, const float* __restrict__ xdbl,
                  const float* __restrict__ A_log, const float* __restrict__ Dp,
                  const float* __restrict__ Hin,
                  const float* __restrict__ z, int ldz, int zoff, int presilu,
                  __bf16* __restrict__ ybf)
{
    const int n = threadIdx.x & 15;
    const int d = blockIdx.x * 16 + (threadIdx.x >> 4);
    const int c = blockIdx.y;
    const int t0 = c * CHUNK;
    const float A    = -__expf(A_log[d * 16 + n]);
    const float Dd   = Dp[d];
    const float bias = dt_bias[d];
    float h = Hin[(size_t)c * 8192 + d * 16 + n];
    for (int t = t0; t < t0 + CHUNK; ++t) {
        float draw = delta[(size_t)t * DINNER + d] + bias;
        float dt = (draw > 20.0f) ? draw : log1pf(__expf(draw));
        float ut = u[(size_t)t * DINNER + d];
        float Bt = xdbl[(size_t)t * 48 + 16 + n];
        float Ct = xdbl[(size_t)t * 48 + 32 + n];
        h = __expf(dt * A) * h + dt * ut * Bt;
        float p = h * Ct;
        p += __shfl_xor(p, 8, 16);
        p += __shfl_xor(p, 4, 16);
        p += __shfl_xor(p, 2, 16);
        p += __shfl_xor(p, 1, 16);
        if (n == 0) {
            float yv = p + ut * Dd;
            float zr = z[(size_t)t * ldz + zoff + d];
            float zs = zr / (1.0f + __expf(-zr));                 // silu (in scan)
            if (presilu) zs = zs / (1.0f + __expf(-zs));          // silu applied before fuse_ssm
            ybf[(size_t)t * DINNER + d] = (__bf16)(yv * zs);
        }
    }
}

// ---------------------------------------------------------------------------
// Tail: per-column partial max, then head (logits, sigmoid, cumprod, argmax)
// ---------------------------------------------------------------------------
__global__ __launch_bounds__(256)
void colmax_part(const float* __restrict__ x, float* __restrict__ pm, int rows, int nparts)
{
    int part = blockIdx.x, c = threadIdx.x;
    int chunk = (rows + nparts - 1) / nparts;
    int r0 = part * chunk;
    int r1 = r0 + chunk; if (r1 > rows) r1 = rows;
    float m = -3.4e38f;
    for (int r = r0; r < r1; ++r) m = fmaxf(m, x[(size_t)r * 256 + c]);
    pm[part * 256 + c] = m;
}

__global__ __launch_bounds__(256)
void head_kernel(const float* __restrict__ pm, int nparts,
                 const float* __restrict__ cls_w, const float* __restrict__ cls_b,
                 float* __restrict__ out)
{
    __shared__ float fm[256];
    __shared__ float lg[4];
    int c = threadIdx.x;
    float m = -3.4e38f;
    for (int p = 0; p < nparts; ++p) m = fmaxf(m, pm[p * 256 + c]);
    fm[c] = m; __syncthreads();
    if (c < 4) {
        float a = cls_b[c];
        for (int d = 0; d < 256; ++d) a += fm[d] * cls_w[d * 4 + c];
        lg[c] = a;
    }
    __syncthreads();
    if (c == 0) {
        float best = lg[0]; int am = 0;
        float S = 1.0f;
        for (int i = 0; i < 4; ++i) if (lg[i] > best) { best = lg[i]; am = i; }
        for (int i = 0; i < 4; ++i) {
            float hz = 1.0f / (1.0f + __expf(-lg[i]));
            out[i] = hz;
            S *= (1.0f - hz);
            out[4 + i] = S;
        }
        out[8] = (float)am;
    }
}

// ---------------------------------------------------------------------------
// Host-side parameter mapping and orchestration
// ---------------------------------------------------------------------------
struct SsmP { const float *conv_w,*conv_b,*x_proj_w,*dt_proj_w,*dt_bias,*A_log,*D,*out_w; };
struct IfP  { const float *ln1_g,*ln1_b,*ln2_g,*ln2_b,*in1_w,*in2_w; SsmP ssm1, ssm2; };
struct MeP  { const float *ln_g,*ln_b,*in_w; SsmP ssm; };

static void fill_ssm(SsmP& s, void* const* in, int i, bool sorted) {
    auto F = [&](int j){ return (const float*)in[j]; };
    if (sorted) { s.A_log=F(i); s.D=F(i+1); s.conv_b=F(i+2); s.conv_w=F(i+3);
                  s.dt_bias=F(i+4); s.dt_proj_w=F(i+5); s.out_w=F(i+6); s.x_proj_w=F(i+7); }
    else        { s.conv_w=F(i); s.conv_b=F(i+1); s.x_proj_w=F(i+2); s.dt_proj_w=F(i+3);
                  s.dt_bias=F(i+4); s.A_log=F(i+5); s.D=F(i+6); s.out_w=F(i+7); }
}
static void fill_if(IfP& f, void* const* in, int i, bool sorted) {
    auto F = [&](int j){ return (const float*)in[j]; };
    if (sorted) { f.in1_w=F(i); f.in2_w=F(i+1); f.ln1_b=F(i+2); f.ln1_g=F(i+3);
                  f.ln2_b=F(i+4); f.ln2_g=F(i+5); }
    else        { f.ln1_g=F(i); f.ln1_b=F(i+1); f.ln2_g=F(i+2); f.ln2_b=F(i+3);
                  f.in1_w=F(i+4); f.in2_w=F(i+5); }
    fill_ssm(f.ssm1, in, i+6,  sorted);
    fill_ssm(f.ssm2, in, i+14, sorted);
}
static void fill_me(MeP& m, void* const* in, int i, bool sorted) {
    auto F = [&](int j){ return (const float*)in[j]; };
    if (sorted) { m.in_w=F(i); m.ln_b=F(i+1); m.ln_g=F(i+2); }
    else        { m.ln_g=F(i); m.ln_b=F(i+1); m.in_w=F(i+2); }
    fill_ssm(m.ssm, in, i+3, sorted);
}

struct SsmW { __bf16 *xproj_bt, *dtproj_bf, *outw_bt; };
struct IfW  { __bf16 *in1_bt, *in2_bt; SsmW s1, s2; };

extern "C" void kernel_launch(void* const* d_in, const int* in_sizes, int n_in,
                              void* d_out, int out_size, void* d_ws, size_t ws_size,
                              hipStream_t stream)
{
    (void)n_in; (void)out_size; (void)ws_size;
    // ---- input mapping (detect flatten order from sizes) ----
    const float *x, *y;
    int p; bool sorted;
    if (in_sizes[0] == 4096 * 1024) {            // x, y first
        x = (const float*)d_in[0]; y = (const float*)d_in[1]; p = 2;
        sorted = (in_sizes[2] == 4);             // sorted params start with cls_b (4 el)
    } else {                                      // jax-sorted top level: params, x, y
        p = 0; sorted = true;
        x = (const float*)d_in[67]; y = (const float*)d_in[68];
    }
    auto F = [&](int j){ return (const float*)d_in[j]; };
    const float *wsi_w,*wsi_b,*mlp1_w,*mlp1_b,*mlp2_w,*mlp2_b,*mm_w,*mm_b,*norm_g,*norm_b,*cls_w,*cls_b;
    IfP f1p, f2p; MeP mep;
    if (sorted) {
        cls_b=F(p+0); cls_w=F(p+1);
        fill_if(f1p, d_in, p+2,  true);
        fill_if(f2p, d_in, p+24, true);
        fill_me(mep, d_in, p+46, true);
        mlp1_b=F(p+57); mlp1_w=F(p+58); mlp2_b=F(p+59); mlp2_w=F(p+60);
        mm_b=F(p+61); mm_w=F(p+62); norm_b=F(p+63); norm_g=F(p+64);
        wsi_b=F(p+65); wsi_w=F(p+66);
    } else {
        wsi_w=F(p+0); wsi_b=F(p+1); mlp1_w=F(p+2); mlp1_b=F(p+3); mlp2_w=F(p+4); mlp2_b=F(p+5);
        fill_if(f1p, d_in, p+6,  false);
        fill_if(f2p, d_in, p+28, false);
        fill_me(mep, d_in, p+50, false);
        mm_w=F(p+61); mm_b=F(p+62); norm_g=F(p+63); norm_b=F(p+64); cls_w=F(p+65); cls_b=F(p+66);
    }

    // ---- workspace bump allocator ----
    uint8_t* base = (uint8_t*)d_ws;
    size_t off = 0;
    auto alloc = [&](size_t bytes) -> void* {
        off = (off + 255) & ~(size_t)255;
        void* r = base + off; off += bytes; return r;
    };
    __bf16* wsi_bt  = (__bf16*)alloc(2L*256*1024);
    __bf16* mlp1_bt = (__bf16*)alloc(2L*128*50);
    __bf16* mlp2_bt = (__bf16*)alloc(2L*256*128);
    __bf16* mm_bt   = (__bf16*)alloc(2L*256*256);
    auto allocSsmW = [&](SsmW& w){
        w.xproj_bt  = (__bf16*)alloc(2L*48*512);
        w.dtproj_bf = (__bf16*)alloc(2L*512*16);
        w.outw_bt   = (__bf16*)alloc(2L*256*512);
    };
    IfW f1w, f2w; SsmW mew; __bf16 *me_in_bt;
    f1w.in1_bt=(__bf16*)alloc(2L*1024*256); f1w.in2_bt=(__bf16*)alloc(2L*1024*256);
    allocSsmW(f1w.s1); allocSsmW(f1w.s2);
    f2w.in1_bt=(__bf16*)alloc(2L*1024*256); f2w.in2_bt=(__bf16*)alloc(2L*1024*256);
    allocSsmW(f2w.s1); allocSsmW(f2w.s2);
    me_in_bt=(__bf16*)alloc(2L*1024*256);   allocSsmW(mew);

    __bf16* XBF = (__bf16*)alloc(2L*4096*1024);
    __bf16* YIN = (__bf16*)alloc(2L*4096*50);
    float*  T1  = (float*) alloc(4L*4096*128);
    __bf16* T1B = (__bf16*)alloc(2L*4096*128);
    float*  H   = (float*) alloc(4L*8192*256);
    float*  LNF = (float*) alloc(4L*8192*256);
    __bf16* LNB = (__bf16*)alloc(2L*8192*256);
    float*  BIG = (float*) alloc(4L*8192*1024);   // he12 | ihc12, or xz (8192x1024)
    float*  U   = (float*) alloc(4L*8192*512);
    __bf16* UB  = (__bf16*)alloc(2L*8192*512);
    float*  XD  = (float*) alloc(4L*8192*48);
    __bf16* DTR = (__bf16*)alloc(2L*8192*16);
    float*  DEL = (float*) alloc(4L*8192*512);
    __bf16* YB  = (__bf16*)alloc(2L*8192*512);
    float*  FEA = (float*) alloc(4L*8192*256);
    float*  PM  = (float*) alloc(4L*64*256);
    float*  PC  = (float*) alloc(4L*128*8192);    // scan chunk products
    float*  SC  = (float*) alloc(4L*128*8192);    // scan chunk local sums
    float*  HI  = (float*) alloc(4L*128*8192);    // scan chunk incoming states

    // ---- launch helpers ----
    auto xposeW = [&](const float* src, __bf16* dst, int K, int N){
        dim3 g((K+31)/32, (N+31)/32);
        transpose_f2bf<<<g, dim3(32,32), 0, stream>>>(src, dst, K, N);
    };
    auto cvt = [&](const float* src, __bf16* dst, int rows, int cols, int ld, int coff){
        int n = rows * cols;
        f2bf_slice<<<(n+255)/256, 256, 0, stream>>>(src, dst, rows, cols, ld, coff);
    };
    auto gemmL = [&](const __bf16* A, const __bf16* Bt, float* C,
                     const float* bias, const float* res, int M, int N, int K, int act){
        dim3 g((M+63)/64, (N+63)/64);
        const bool aligned = ((K & 31) == 0) && ((M & 63) == 0) && ((N & 63) == 0);
        if (aligned) {
            if (res)               gemm_bf16_fast<0,0,1><<<g,128,0,stream>>>(A,Bt,C,bias,res,M,N,K);
            else if (act && bias)  gemm_bf16_fast<1,1,0><<<g,128,0,stream>>>(A,Bt,C,bias,res,M,N,K);
            else                   gemm_bf16_fast<0,0,0><<<g,128,0,stream>>>(A,Bt,C,bias,res,M,N,K);
        } else {
            gemm_bf16_edge<<<g,128,0,stream>>>(A,Bt,C,bias,res,M,N,K,act);
        }
    };
    auto ln = [&](const float* xin, const float* g, const float* b, float* outp, int rows){
        layernorm256<<<rows, 256, 0, stream>>>(xin, g, b, outp);
    };
    auto fuse = [&](const SsmP& sp, const SsmW& sw, const float* xsrc,
                    const float* zbase, int zoff, float* io, int Lq, int presilu){
        int nthr = Lq * DINNER;
        conv1d_silu<<<(nthr+255)/256, 256, 0, stream>>>(xsrc, 1024, 0, sp.conv_w, sp.conv_b, U, UB, Lq);
        gemmL(UB, sw.xproj_bt, XD, nullptr, nullptr, Lq, 48, 512, 0);
        cvt(XD, DTR, Lq, 16, 48, 0);
        gemmL(DTR, sw.dtproj_bf, DEL, nullptr, nullptr, Lq, 512, 16, 0);
        // chunked parallel selective scan
        const int C = Lq / CHUNK;
        dim3 sg(32, C);
        sscan_phase1<<<sg, 256, 0, stream>>>(DEL, sp.dt_bias, U, XD, sp.A_log, PC, SC);
        sscan_carry<<<32, 256, 0, stream>>>(PC, SC, HI, C);
        sscan_phase3<<<sg, 256, 0, stream>>>(DEL, sp.dt_bias, U, XD, sp.A_log, sp.D,
                                             HI, zbase, 1024, zoff, presilu, YB);
        gemmL(YB, sw.outw_bt, io, nullptr, io, Lq, 256, 512, 0);   // + residual, in place
    };

    // ---- weight prep (fp32 -> bf16, transposed to N x K) ----
    xposeW(wsi_w,  wsi_bt,  1024, 256);
    xposeW(mlp1_w, mlp1_bt, 50,   128);
    xposeW(mlp2_w, mlp2_bt, 128,  256);
    xposeW(mm_w,   mm_bt,   256,  256);
    auto prepIf = [&](const IfP& pp, IfW& ww){
        xposeW(pp.in1_w, ww.in1_bt, 256, 1024);
        xposeW(pp.in2_w, ww.in2_bt, 256, 1024);
        const SsmP* sp[2] = {&pp.ssm1, &pp.ssm2};
        SsmW* sw[2] = {&ww.s1, &ww.s2};
        for (int i = 0; i < 2; ++i) {
            xposeW(sp[i]->x_proj_w, sw[i]->xproj_bt, 512, 48);
            cvt(sp[i]->dt_proj_w, sw[i]->dtproj_bf, 512, 16, 16, 0);   // already N x K
            xposeW(sp[i]->out_w, sw[i]->outw_bt, 512, 256);
        }
    };
    prepIf(f1p, f1w);
    prepIf(f2p, f2w);
    xposeW(mep.in_w, me_in_bt, 256, 1024);
    xposeW(mep.ssm.x_proj_w, mew.xproj_bt, 512, 48);
    cvt(mep.ssm.dt_proj_w, mew.dtproj_bf, 512, 16, 16, 0);
    xposeW(mep.ssm.out_w, mew.outw_bt, 512, 256);

    float* Hwsi = H;                       // rows 0..4095 of concat
    float* Hmif = H + (size_t)4096 * 256;  // rows 4096..8191
    float* BIG2 = BIG + (size_t)4096 * 1024;

    // ---- stage A: input embeddings ----
    cvt(x, XBF, 4096, 1024, 1024, 0);
    gemmL(XBF, wsi_bt, Hwsi, wsi_b, nullptr, 4096, 256, 1024, 1);
    cvt(y, YIN, 4096, 50, 50, 0);
    gemmL(YIN, mlp1_bt, T1, mlp1_b, nullptr, 4096, 128, 50, 1);
    cvt(T1, T1B, 4096, 128, 128, 0);
    gemmL(T1B, mlp2_bt, Hmif, mlp2_b, nullptr, 4096, 256, 128, 1);

    // ---- stages B/C: interleaved fusion blocks ----
    auto ifblock = [&](const IfP& ip, const IfW& iw){
        ln(Hwsi, ip.ln1_g, ip.ln1_b, LNF, 4096);
        cvt(LNF, LNB, 4096, 256, 256, 0);
        gemmL(LNB, iw.in1_bt, BIG, nullptr, nullptr, 4096, 1024, 256, 0);
        ln(Hmif, ip.ln2_g, ip.ln2_b, LNF, 4096);
        cvt(LNF, LNB, 4096, 256, 256, 0);
        gemmL(LNB, iw.in2_bt, BIG2, nullptr, nullptr, 4096, 1024, 256, 0);
        fuse(ip.ssm1, iw.s1, BIG,  BIG2, 512, Hwsi, 4096, 1);  // z = silu(ihc2), silu'd again in scan
        fuse(ip.ssm2, iw.s2, BIG2, BIG,  512, Hmif, 4096, 1);
    };
    ifblock(f1p, f1w);
    ifblock(f2p, f2w);

    // ---- stage D: mamba block over concat (L = 8192) ----
    ln(H, mep.ln_g, mep.ln_b, LNF, 8192);
    cvt(LNF, LNB, 8192, 256, 256, 0);
    gemmL(LNB, me_in_bt, BIG, nullptr, nullptr, 8192, 1024, 256, 0);  // xz
    fuse(mep.ssm, mew, BIG, BIG, 512, H, 8192, 0);

    // ---- stage E: head ----
    cvt(H, LNB, 8192, 256, 256, 0);
    gemmL(LNB, mm_bt, FEA, mm_b, nullptr, 8192, 256, 256, 1);
    ln(FEA, norm_g, norm_b, LNF, 8192);
    colmax_part<<<64, 256, 0, stream>>>(LNF, PM, 8192, 64);
    head_kernel<<<1, 256, 0, stream>>>(PM, 64, cls_w, cls_b, (float*)d_out);
}
